// DualStreamAttentionWithRotaryEmbedding_29008209117604
// MI455X (gfx1250) — compile-verified, matrix operand order checked by reading, not run on hardware
//
#include <hip/hip_runtime.h>
#include <hip/hip_bf16.h>

typedef __attribute__((ext_vector_type(16))) __bf16 v16bf;
typedef __attribute__((ext_vector_type(8)))  float  v8f;
typedef __attribute__((ext_vector_type(4)))  unsigned int v4u;
typedef __attribute__((ext_vector_type(8)))  int v8i;
typedef __attribute__((ext_vector_type(4)))  int v4i;

namespace {
constexpr int B  = 2;
constexpr int L  = 2048;
constexpr int S  = 512;
constexpr int D  = 2048;
constexpr int H  = 16;
constexpr int HD = D / H;      // 128
constexpr int T  = S + L;      // 2560
constexpr int BT = B * T;      // 5120
constexpr int D3 = 3 * D;      // 6144
constexpr float EPS   = 1e-6f;
constexpr float SCALE = 0.08838834764831845f; // 1/sqrt(HD)

constexpr int QBLK = 2048;     // bf16 per 16-row Q fragment block (4*32*16)
constexpr int KBLK = 4096;     // bf16 per 32-key K (or V) fragment block (8*32*16)

__device__ __forceinline__ __bf16 f2bf(float f) {
  union { float f; unsigned u; } v; v.f = f;
  unsigned r = v.u + 0x7FFFu + ((v.u >> 16) & 1u);   // round-to-nearest-even
  union { unsigned short s; __bf16 b; } o; o.s = (unsigned short)(r >> 16);
  return o.b;
}

// CDNA5 16-bit WMMA operand layout (ISA 7.12.2): element at K-index kd,
// "row index within 16" rn (A: M row; B: N column) lives in
//   lane = rn + 16*((kd>>3)&1),  slot = 2*(((kd&16)>>2) + ((kd&7)>>1)) + (kd&1)
__device__ __forceinline__ int fragLane(int rn, int kd) {
  return rn + 16 * ((kd >> 3) & 1);
}
__device__ __forceinline__ int fragSlot(int kd) {
  return 2 * (((kd & 16) >> 2) + ((kd & 7) >> 1)) + (kd & 1);
}

__device__ __forceinline__ v16bf ldsFrag(const __bf16* p) {
  return *(const v16bf*)p;   // 32B contiguous, 32B aligned -> 2x ds_load_b128
}

#if __has_builtin(__builtin_amdgcn_tensor_load_to_lds)
#define HAVE_TDM 1
// TDM DMA of one contiguous 8 KB block (1024 x 8B) from global to LDS.
// D# per CDNA5 ISA ch.8: group0 = {count|flags, lds_addr, global_addr_lo,
// global_addr_hi | type=2}; group1 packs data_size=3 (8B), tensor_dim0=1024,
// tile_dim0=1024, tensor_dim0_stride=1024, workgroup_mask=0 (no cluster).
__device__ __forceinline__ void tdm_load_8kb(const void* gptr, unsigned lds_off) {
  unsigned long long ga = (unsigned long long)(uintptr_t)gptr;
  v4u g0 = { 1u,                                        // count=1 (user descriptor)
             lds_off,                                    // lds_addr (bytes)
             (unsigned)(ga & 0xFFFFFFFFu),               // global_addr[31:0]
             (unsigned)((ga >> 32) & 0x1FFFFFFu) | (2u << 30) };  // [56:32] | type=2
  v8i g1 = { 0x30000,                 // [17:16] data_size=3 (8 bytes)
             (int)(1024u << 16),      // [79:48] tensor_dim0 = 1024 (lo half)
             0x00010000,              // [111:80] tensor_dim1 = 1
             (int)(1024u << 16),      // [127:112] tile_dim0 = 1024
             0,                       // tile_dim1/2 unused
             1024,                    // [207:160] tensor_dim0_stride = 1024
             0, 0 };
  v4i gz4 = { 0, 0, 0, 0 };
  v8i gz8 = { 0, 0, 0, 0, 0, 0, 0, 0 };
  __builtin_amdgcn_tensor_load_to_lds(g0, g1, gz4, gz4, gz8, 0);
}
#else
#define HAVE_TDM 0
#endif
} // namespace

// -------------------------------------------------------------------------
// Kernel 1: fused QK and V projections for both streams.
// qkvF[BT][3D] fp32: cols [0,2D) = Wqk output, [2D,3D) = Wv output (+bias).
// Block: 128 threads (4 waves), 16x64 tile, K-step 32.
// -------------------------------------------------------------------------
__global__ __launch_bounds__(128) void qkv_proj_kernel(
    const float* __restrict__ x, const float* __restrict__ c,
    const float* __restrict__ Wqk_x, const float* __restrict__ Wv_x, const float* __restrict__ bv_x,
    const float* __restrict__ Wqk_c, const float* __restrict__ Wv_c, const float* __restrict__ bv_c,
    float* __restrict__ qkv)
{
  __shared__ alignas(32) __bf16 As[32][16];        // A 16x32, fragment-major
  __shared__ alignas(32) __bf16 Bs[4][32][16];     // B 32x64 -> 4 wave tiles, fragment-major

  const int tid  = threadIdx.x;
  const int lane = tid & 31;
  const int wave = tid >> 5;
  const int half = lane >> 4;
  const int ln16 = lane & 15;

  const int r0   = blockIdx.x * 16;   // global token row
  const int col0 = blockIdx.y * 64;   // column in [0, 3D)
  const int b    = r0 / T;
  const int t0   = r0 % T;
  const bool fromC = (t0 < S);

  const float* src = fromC ? (c + ((size_t)b * S + t0) * D)
                           : (x + ((size_t)b * L + (t0 - S)) * D);

  const float* W; int ldw; const float* bias = nullptr; int wcol;
  if (col0 < 2 * D) { W = fromC ? Wqk_c : Wqk_x; ldw = 2 * D; wcol = col0; }
  else              { W = fromC ? Wv_c  : Wv_x;  ldw = D;     wcol = col0 - 2 * D;
                      bias = fromC ? bv_c : bv_x; }

  // Per-thread staging assignments (loop-invariant LDS scatter addresses).
  const int am  = tid >> 5, akd = tid & 31;        // A rows am+4u, feature akd
  const float* aSrc = src + (size_t)am * D + akd;
  __bf16* aDst[4];
#pragma unroll
  for (int u = 0; u < 4; ++u)
    aDst[u] = &As[fragLane(am + 4 * u, akd)][fragSlot(akd)];

  const int bn = tid & 63, bk0 = tid >> 6;         // B rows bk0+2u, column bn
  const float* bSrc = W + (size_t)bk0 * ldw + wcol + bn;
  __bf16* bDst[16];
#pragma unroll
  for (int u = 0; u < 16; ++u)
    bDst[u] = &Bs[bn >> 4][fragLane(bn & 15, bk0 + 2 * u)][fragSlot(bk0 + 2 * u)];

  v8f acc = {};

  for (int kk = 0; kk < D; kk += 32) {
    __syncthreads();
#pragma unroll
    for (int u = 0; u < 4; ++u)
      *aDst[u] = f2bf(aSrc[(size_t)(4 * u) * D + kk]);
#pragma unroll
    for (int u = 0; u < 16; ++u)
      *bDst[u] = f2bf(bSrc[(size_t)(kk + 2 * u) * ldw]);
    __syncthreads();

    v16bf a  = ldsFrag(As[lane]);
    v16bf bf = ldsFrag(Bs[wave][lane]);
    acc = __builtin_amdgcn_wmma_f32_16x16x32_bf16(false, a, false, bf,
                                                  (short)0, acc, false, false);
  }

  const int   colw = col0 + wave * 16 + ln16;
  const float bvv  = bias ? bias[wcol + wave * 16 + ln16] : 0.0f;
#pragma unroll
  for (int j = 0; j < 8; ++j) {
    int m = j + 8 * half;                       // C/D layout: row = reg + 8*lane-half
    qkv[(size_t)(r0 + m) * D3 + colw] = acc[j] + bvv;
  }
}

// -------------------------------------------------------------------------
// Kernel 2: per-head RMSNorm on q,k + RoPE, then scatter q/k/v as bf16 into
// WMMA-fragment-ordered global buffers for the attention kernel.
// grid = (BT, H), block = 128 (= HD)
// -------------------------------------------------------------------------
__global__ __launch_bounds__(128) void normrope_kernel(
    const float* __restrict__ qkv, const float* __restrict__ freqs,
    const float* __restrict__ qn_x, const float* __restrict__ kn_x,
    const float* __restrict__ qn_c, const float* __restrict__ kn_c,
    __bf16* __restrict__ Qb, __bf16* __restrict__ Kb, __bf16* __restrict__ Vb)
{
  const int tok = blockIdx.x;
  const int h   = blockIdx.y;
  const int d   = threadIdx.x;
  const int bb  = tok / T;
  const int t   = tok % T;
  const bool fromC = (t < S);
  const float* qn = fromC ? qn_c : qn_x;
  const float* kn = fromC ? kn_c : kn_x;

  const float* qrow = qkv + (size_t)tok * D3 + h * HD;
  const float* krow = qrow + D;
  const float* vrow = qrow + 2 * D;
  float q = qrow[d], k = krow[d], v = vrow[d];

  float sq = q * q, sk = k * k;
#pragma unroll
  for (int off = 16; off; off >>= 1) {
    sq += __shfl_xor(sq, off, 32);
    sk += __shfl_xor(sk, off, 32);
  }
  __shared__ float red[4][2];
  if ((threadIdx.x & 31) == 0) { red[threadIdx.x >> 5][0] = sq; red[threadIdx.x >> 5][1] = sk; }
  __syncthreads();
  float tq = red[0][0] + red[1][0] + red[2][0] + red[3][0];
  float tk = red[0][1] + red[1][1] + red[2][1] + red[3][1];

  q = q * rsqrtf(tq * (1.0f / HD) + EPS) * qn[d];
  k = k * rsqrtf(tk * (1.0f / HD) + EPS) * kn[d];

  // RoPE over interleaved pairs (2i, 2i+1): adjacent lanes exchange partners.
  float qo = __shfl_xor(q, 1, 32);
  float ko = __shfl_xor(k, 1, 32);
  const float* fc = freqs + (((size_t)bb * T + t) * (HD / 2) + (d >> 1)) * 2;
  float cs = fc[0], sn = fc[1];
  float qr = (d & 1) ? (qo * sn + q * cs) : (q * cs - qo * sn);
  float kr = (d & 1) ? (ko * sn + k * cs) : (k * cs - ko * sn);

  const int bh  = bb * H + h;
  const int kk  = d >> 5, kdl = d & 31;

  // Q -> A-fragment order, blocks of 16 query rows.
  size_t qoff = ((size_t)bh * (T / 16) + (t >> 4)) * QBLK;
  Qb[qoff + ((size_t)(kk * 32) + fragLane(t & 15, kdl)) * 16 + fragSlot(kdl)] = f2bf(qr);

  // K -> score B-fragment order, blocks of 32 keys: subtile nt=(t%32)/16, chunk kk.
  const int kw = t & 31;
  size_t koff = ((size_t)bh * (T / 32) + (t >> 5)) * KBLK;
  Kb[koff + ((size_t)((kw >> 4) * 4 + kk) * 32 + fragLane(kw & 15, kdl)) * 16
          + fragSlot(kdl)] = f2bf(kr);

  // V -> PV B-fragment order: B[kd=key%32][n=d%16], column tile cidx=d/16.
  Vb[koff + ((size_t)(d >> 4) * 32 + fragLane(d & 15, kw)) * 16 + fragSlot(kw)] = f2bf(v);
}

// -------------------------------------------------------------------------
// Kernel 3: flash attention per (b, h); block = 128 query rows (8 waves x 16),
// streaming 32-key fragment-ordered K/V blocks into LDS via the Tensor Data
// Mover (TENSOR_LOAD_TO_LDS + s_wait_tensorcnt), one issue per block.
// grid = (T/128, H, B), block = 256
// -------------------------------------------------------------------------
__global__ __launch_bounds__(256) void attn_kernel(
    const __bf16* __restrict__ Qb, const __bf16* __restrict__ Kb,
    const __bf16* __restrict__ Vb, float* __restrict__ out)
{
  __shared__ alignas(32) __bf16 Ks[8][32][16];   // [nt*4+kk][lane][slot], 8 KB
  __shared__ alignas(32) __bf16 Vs[8][32][16];   // [cidx]   [lane][slot], 8 KB
  __shared__ alignas(32) __bf16 Ps[8][32][16];   // [wave]   [lane][slot], 8 KB

  const int tid  = threadIdx.x;
  const int lane = tid & 31;
  const int wave = tid >> 5;
  const int half = lane >> 4;
  const int ln16 = lane & 15;

  const int h  = blockIdx.y;
  const int b  = blockIdx.z;
  const int bh = b * H + h;
  const int q0 = blockIdx.x * 128 + wave * 16;

  // Q fragments: contiguous 32B vector loads, already in A-fragment order.
  const __bf16* qbase = Qb + ((size_t)bh * (T / 16) + (q0 >> 4)) * QBLK;
  v16bf qf[4];
#pragma unroll
  for (int kk = 0; kk < 4; ++kk)
    qf[kk] = *(const v16bf*)(qbase + ((size_t)(kk * 32) + lane) * 16);

  const v8f zero8 = {};
  v8f o[8];
  float rowmax[8], rowsum[8];
#pragma unroll
  for (int j = 0; j < 8; ++j) { o[j] = zero8; rowmax[j] = -3.0e38f; rowsum[j] = 0.0f; }

  const __bf16* gkBase = Kb + (size_t)bh * (T / 32) * KBLK;
  const __bf16* gvBase = Vb + (size_t)bh * (T / 32) * KBLK;
#if HAVE_TDM
  const unsigned ksOff = (unsigned)(uintptr_t)&Ks[0][0][0];
  const unsigned vsOff = (unsigned)(uintptr_t)&Vs[0][0][0];
#endif

  for (int kt = 0; kt < T; kt += 32) {
    const __bf16* gk = gkBase + (size_t)(kt >> 5) * KBLK;
    const __bf16* gv = gvBase + (size_t)(kt >> 5) * KBLK;
    __syncthreads();                       // previous tile fully consumed
#if HAVE_TDM
    if (wave == 0) {                       // TDM ignores EXEC; one issue per block
      tdm_load_8kb(gk, ksOff);
      tdm_load_8kb(gv, vsOff);
      __builtin_amdgcn_s_wait_tensorcnt(0);
    }
#else
    {
      const uint4* g4k = (const uint4*)gk;
      const uint4* g4v = (const uint4*)gv;
      uint4* lk = (uint4*)&Ks[0][0][0];
      uint4* lv = (uint4*)&Vs[0][0][0];
#pragma unroll
      for (int u = 0; u < 2; ++u) {
        lk[tid + 256 * u] = g4k[tid + 256 * u];
        lv[tid + 256 * u] = g4v[tid + 256 * u];
      }
    }
#endif
    // Prefetch next key block into cache, one probe per 64B line.
    if (kt + 32 < T && tid < 128) {
      __builtin_prefetch((const char*)gk + 8192 + tid * 64);
      __builtin_prefetch((const char*)gv + 8192 + tid * 64);
    }
    __syncthreads();                       // publish LDS tile to all waves

    // scores (16 x 32): two 16x16 column tiles, K-depth 128 = 4 WMMAs each
    v8f s[2]; s[0] = zero8; s[1] = zero8;
#pragma unroll
    for (int nt = 0; nt < 2; ++nt) {
#pragma unroll
      for (int kk = 0; kk < 4; ++kk) {
        v16bf bf = ldsFrag(Ks[nt * 4 + kk][lane]);
        s[nt] = __builtin_amdgcn_wmma_f32_16x16x32_bf16(false, qf[kk], false, bf,
                                                        (short)0, s[nt], false, false);
      }
    }

    // Online softmax; row (j + 8*half) lives on the 16 lanes of this half.
#pragma unroll
    for (int j = 0; j < 8; ++j) {
      float v0 = s[0][j] * SCALE;
      float v1 = s[1][j] * SCALE;
      float m = fmaxf(v0, v1);
#pragma unroll
      for (int off = 1; off < 16; off <<= 1) m = fmaxf(m, __shfl_xor(m, off, 32));
      float mnew  = fmaxf(rowmax[j], m);
      float alpha = __expf(rowmax[j] - mnew);
      rowmax[j] = mnew;
      float p0 = __expf(v0 - mnew);
      float p1 = __expf(v1 - mnew);
      float ps = p0 + p1;
#pragma unroll
      for (int off = 1; off < 16; off <<= 1) ps += __shfl_xor(ps, off, 32);
      rowsum[j] = rowsum[j] * alpha + ps;
#pragma unroll
      for (int cidx = 0; cidx < 8; ++cidx) o[cidx][j] *= alpha;
      // Scatter P(row m, key kd) straight into A-fragment order.
      int m0  = j + 8 * half;
      int kd0 = ln16;              // p0: keys 0..15
      int kd1 = 16 + ln16;         // p1: keys 16..31
      Ps[wave][fragLane(m0, kd0)][fragSlot(kd0)] = f2bf(p0);
      Ps[wave][fragLane(m0, kd1)][fragSlot(kd1)] = f2bf(p1);
    }
    // Per-wave staging buffer: drain the DS counter, no cross-wave barrier needed.
    asm volatile("s_wait_dscnt 0x0" ::: "memory");

    // P (16x32) as A-fragment, O += P @ V (8 x 16-col tiles of HD)
    v16bf pf = ldsFrag(Ps[wave][lane]);
#pragma unroll
    for (int cidx = 0; cidx < 8; ++cidx) {
      v16bf vf = ldsFrag(Vs[cidx][lane]);
      o[cidx] = __builtin_amdgcn_wmma_f32_16x16x32_bf16(false, pf, false, vf,
                                                        (short)0, o[cidx], false, false);
    }
  }

  // Epilogue: 1/rowsum, scatter into y_x / y_c regions of d_out.
#pragma unroll
  for (int j = 0; j < 8; ++j) rowsum[j] = 1.0f / rowsum[j];
#pragma unroll
  for (int cidx = 0; cidx < 8; ++cidx) {
#pragma unroll
    for (int j = 0; j < 8; ++j) {
      int t   = q0 + j + 8 * half;
      int col = h * HD + cidx * 16 + ln16;
      float v = o[cidx][j] * rowsum[j];
      float* dst;
      if (t < S) dst = out + (size_t)B * L * D + ((size_t)b * S + t) * D + col;        // y_c
      else       dst = out + ((size_t)b * L + (t - S)) * D + col;                      // y_x
      *dst = v;
    }
  }
}

// -------------------------------------------------------------------------
extern "C" void kernel_launch(void* const* d_in, const int* in_sizes, int n_in,
                              void* d_out, int out_size, void* d_ws, size_t ws_size,
                              hipStream_t stream) {
  (void)in_sizes; (void)n_in; (void)out_size; (void)ws_size;
  const float* x      = (const float*)d_in[0];
  const float* c      = (const float*)d_in[1];
  const float* freqs  = (const float*)d_in[2];
  const float* Wqk_x  = (const float*)d_in[3];
  const float* Wv_x   = (const float*)d_in[4];
  const float* bv_x   = (const float*)d_in[5];
  const float* qn_x   = (const float*)d_in[6];
  const float* kn_x   = (const float*)d_in[7];
  const float* Wqk_c  = (const float*)d_in[8];
  const float* Wv_c   = (const float*)d_in[9];
  const float* bv_c   = (const float*)d_in[10];
  const float* qn_c   = (const float*)d_in[11];
  const float* kn_c   = (const float*)d_in[12];

  float*  qkvF = (float*)d_ws;                                    // [BT][3D] fp32
  __bf16* Qb   = (__bf16*)(qkvF + (size_t)BT * D3);               // 20 MB
  __bf16* Kb   = Qb + (size_t)B * H * (T / 16) * QBLK;            // 20 MB
  __bf16* Vb   = Kb + (size_t)B * H * (T / 32) * KBLK;            // 20 MB
  float*  out  = (float*)d_out;

  dim3 g1(BT / 16, D3 / 64);
  qkv_proj_kernel<<<g1, 128, 0, stream>>>(x, c, Wqk_x, Wv_x, bv_x,
                                          Wqk_c, Wv_c, bv_c, qkvF);

  dim3 g2(BT, H);
  normrope_kernel<<<g2, 128, 0, stream>>>(qkvF, freqs, qn_x, kn_x, qn_c, kn_c,
                                          Qb, Kb, Vb);

  dim3 g3(T / 128, H, B);
  attn_kernel<<<g3, 256, 0, stream>>>(Qb, Kb, Vb, out);
}
